// EventMamba_88476326297813
// MI455X (gfx1250) — compile-verified, matrix operand order checked by reading, not run on hardware
//
#include <hip/hip_runtime.h>
#include <math.h>

typedef __attribute__((ext_vector_type(16))) _Float16 v16h;
typedef __attribute__((ext_vector_type(8)))  _Float16 v8h;
typedef __attribute__((ext_vector_type(8)))  float    v8f;
typedef __attribute__((ext_vector_type(4)))  unsigned int v4u;
typedef __attribute__((ext_vector_type(8)))  int      v8i;
typedef __attribute__((ext_vector_type(4)))  int      v4i;

// ---------------------------------------------------------------------------
// Tensor Data Mover: DMA a 64-row x 32-col f32 tile (row stride = Cstride
// elements) from global into LDS. D# built per CDNA5 ISA §8.3/8.4.
// tensor_dim0/1 are the REMAINING cols/rows from the tile origin so the TDM
// zero-fills out-of-bounds tile elements for us.
// Issued by one wave; completion via s_wait_tensorcnt.
// ---------------------------------------------------------------------------
__device__ inline void tdm_load_tile_2d(const float* gptr, unsigned lds_byte_addr,
                                        int rows_rem, int cols_rem, int Cstride)
{
    unsigned long long ga = (unsigned long long)(size_t)gptr;
    unsigned td0 = (unsigned)cols_rem;   // tensor extent dim0 (elements)
    unsigned td1 = (unsigned)rows_rem;   // tensor extent dim1 (rows)
    unsigned st0 = (unsigned)Cstride;    // dim0 stride (elements)
    v4u g0;
    g0[0] = 1u;                                              // count=1, user D#
    g0[1] = lds_byte_addr;                                   // LDS dest (bytes)
    g0[2] = (unsigned)(ga & 0xffffffffu);                    // global_addr lo
    g0[3] = (unsigned)((ga >> 32) & 0x01ffffffu) | (2u << 30); // addr hi | type=2
    v8i g1;
    g1[0] = (int)(2u << 16);                                 // data_size = 4 bytes
    g1[1] = (int)((td0 & 0xffffu) << 16);                    // tensor_dim0[15:0]
    g1[2] = (int)((td0 >> 16) | ((td1 & 0xffffu) << 16));    // td0 hi | td1 lo
    g1[3] = (int)((td1 >> 16) | (32u << 16));                // td1 hi | tile_dim0=32
    g1[4] = 64;                                              // tile_dim1=64, tile_dim2=0
    g1[5] = (int)st0;                                        // tensor_dim0_stride lo32
    g1[6] = 0;                                               // stride0 hi | stride1 lo
    g1[7] = 0;
    v4i zz4 = {0, 0, 0, 0};
    v8i zz8 = {0, 0, 0, 0, 0, 0, 0, 0};
    __builtin_amdgcn_tensor_load_to_lds(g0, g1, zz4, zz4, zz8, 0);
}

// ---------------------------------------------------------------------------
// Generic WMMA GEMM:  Y[o, m] = act( sum_c W[o,c] * X[c,m] + bias[o] )
//   W: [O x Cin] row-major f32; X: inmode==0 (Bdim,Cin,S) channel-major,
//   inmode==1 channels-last. act: 0 none, 1 relu, 2 silu, 3 softplus.
// Block = 256 threads (8 waves), tile 64(O) x 64(M); wave w computes rows
// (w&3)*16 x cols (w>>2)*32: one A fragment, two B fragments, two accs =>
// 2 x v_wmma_f32_16x16x32_f16 per K-step per wave.
// Staging: W (and X when channels-last) arrive via TDM tensor_load_to_lds
// into f32 LDS tiles, then all threads convert to f16 fragment-layout tiles
// (row stride 40 halfs = 80B so each fragment is two ds_load_b128).
// Mode-0 X is staged by clamped vector loads (batch-straddling layout).
// ---------------------------------------------------------------------------
__global__ __launch_bounds__(256)
void wmma_gemm_kernel(const float* __restrict__ W, const float* __restrict__ X,
                      const float* __restrict__ bias, float* __restrict__ Y,
                      int O, int Cin, int M, int S, int inmode, int outmode, int act)
{
    __shared__ __align__(16) float    sWf[64][32];  // TDM landing pad (W)
    __shared__ __align__(16) float    sXf[64][32];  // TDM landing pad (X, mode1)
    __shared__ __align__(16) _Float16 sW[64][40];   // [row][k] fragment layout
    __shared__ __align__(16) _Float16 sXt[64][40];  // [col][k] fragment layout

    const int tid  = threadIdx.x;
    const int lane = tid & 31;
    const int wave = tid >> 5;            // 0..7
    const int wr   = (wave & 3) * 16;
    const int wc   = (wave >> 2) * 32;
    const int tileM = blockIdx.x * 64;
    const int tileO = blockIdx.y * 64;

    const int l15 = lane & 15;
    const int kb  = (lane < 16) ? 0 : 8;  // ISA 16-bit matrix K split (7.12.2)

    const int kc = tid & 31;              // k offset within tile
    const int r0 = tid >> 5;              // rows/cols handled: r0 + 8j
    const long xs = (inmode == 0) ? (long)S : 1;

    // mode-0 X staging bases (fixed across K loop)
    long xbase[8]; bool xok[8];
    if (inmode == 0) {
        #pragma unroll
        for (int j = 0; j < 8; ++j) {
            int gm = tileM + r0 + 8 * j;
            xok[j] = (gm < M);
            int gmc = xok[j] ? gm : (M - 1);
            xbase[j] = (long)(gmc / S) * Cin * S + (gmc % S);
        }
    }

    v8f acc0 = {};
    v8f acc1 = {};

    for (int k0 = 0; k0 < Cin; k0 += 32) {
        // ---- async tensor DMA of f32 tiles (wave 0 only) ----
        if (wave == 0) {
            tdm_load_tile_2d(W + (long)tileO * Cin + k0,
                             (unsigned)(size_t)&sWf[0][0],
                             O - tileO, Cin - k0, Cin);
            if (inmode == 1)
                tdm_load_tile_2d(X + (long)tileM * Cin + k0,
                                 (unsigned)(size_t)&sXf[0][0],
                                 M - tileM, Cin - k0, Cin);
            __builtin_amdgcn_s_wait_tensorcnt(0);
        }
        // ---- mode-0 X: clamped vector loads straight to f16 tile ----
        if (inmode == 0) {
            int gk = k0 + kc;
            bool kok = (gk < Cin);
            int gkc = kok ? gk : (Cin - 1);
            if (k0 + 32 < Cin)  // speculative prefetch of next K tile
                __builtin_prefetch(X + xbase[0] + (long)(gkc + 32) * xs, 0, 1);
            float xv[8];
            #pragma unroll
            for (int j = 0; j < 8; ++j) xv[j] = X[xbase[j] + (long)gkc * xs];
            #pragma unroll
            for (int j = 0; j < 8; ++j) xv[j] = (xok[j] && kok) ? xv[j] : 0.0f;
            #pragma unroll
            for (int j = 0; j < 8; ++j) sXt[r0 + 8 * j][kc] = (_Float16)xv[j];
        }
        __syncthreads();   // A: TDM tiles visible workgroup-wide

        // ---- convert f32 staging tiles into f16 fragment-layout tiles ----
        #pragma unroll
        for (int j = 0; j < 8; ++j)
            sW[r0 + 8 * j][kc] = (_Float16)sWf[r0 + 8 * j][kc];
        if (inmode == 1) {
            #pragma unroll
            for (int j = 0; j < 8; ++j)
                sXt[r0 + 8 * j][kc] = (_Float16)sXf[r0 + 8 * j][kc];
        }
        __syncthreads();   // B: f16 tiles ready

        // ---- fragments: two contiguous 16B LDS reads each ----
        union { v16h v; v8h h[2]; } a, b0, b1;
        a.h[0]  = *(const v8h*)&sW [wr + l15][kb];
        a.h[1]  = *(const v8h*)&sW [wr + l15][16 + kb];
        b0.h[0] = *(const v8h*)&sXt[wc + l15][kb];
        b0.h[1] = *(const v8h*)&sXt[wc + l15][16 + kb];
        b1.h[0] = *(const v8h*)&sXt[wc + 16 + l15][kb];
        b1.h[1] = *(const v8h*)&sXt[wc + 16 + l15][16 + kb];

        acc0 = __builtin_amdgcn_wmma_f32_16x16x32_f16(
            false, a.v, false, b0.v, (short)0, acc0, false, false);
        acc1 = __builtin_amdgcn_wmma_f32_16x16x32_f16(
            false, a.v, false, b1.v, (short)0, acc1, false, false);
        __syncthreads();   // protect tiles before next iteration's writes
    }

    // ---- epilogue: bias + activation + layout-aware store ----
    #pragma unroll
    for (int r = 0; r < 8; ++r) {
        int row = tileO + wr + r + ((lane >= 16) ? 8 : 0);
        if (row >= O) continue;
        float bval = bias ? bias[row] : 0.0f;
        #pragma unroll
        for (int half = 0; half < 2; ++half) {
            int col = tileM + wc + half * 16 + l15;
            if (col >= M) continue;
            float v = (half == 0) ? acc0[r] : acc1[r];
            v += bval;
            if (act == 1)      v = fmaxf(v, 0.0f);
            else if (act == 2) v = v / (1.0f + __expf(-v));
            else if (act == 3) v = (v > 20.0f) ? v : __logf(1.0f + __expf(v));
            long addr = (outmode == 0)
                ? (long)(col / S) * O * S + (long)row * S + (col % S)
                : (long)col * O + row;
            Y[addr] = v;
        }
    }
}

// ---------------------------------------------------------------------------
// Elementwise / utility kernels
// ---------------------------------------------------------------------------
__global__ void zero_kernel(float* __restrict__ p, long n)
{
    long t = (long)blockIdx.x * blockDim.x + threadIdx.x;
    if (t < n) p[t] = 0.0f;
}

__global__ void add_kernel(const float* __restrict__ a, const float* __restrict__ b,
                           float* __restrict__ c, long n)
{
    long t = (long)blockIdx.x * blockDim.x + threadIdx.x;
    if (t < n) c[t] = a[t] + b[t];
}

__global__ void transpose_in_kernel(const float* __restrict__ x, float* __restrict__ xyz,
                                    int B, int N)
{
    long t = (long)blockIdx.x * blockDim.x + threadIdx.x;
    if (t >= (long)B * N * 3) return;
    int c = (int)(t % 3);
    long sn = t / 3;
    int s = (int)(sn % N);
    int b = (int)(sn / N);
    xyz[t] = x[((long)b * 3 + c) * N + s];
}

__global__ void split_kernel(const float* __restrict__ in, float* __restrict__ out,
                             long M, int Win, int coff, int w)
{
    long t = (long)blockIdx.x * blockDim.x + threadIdx.x;
    if (t >= M * w) return;
    long m = t / w;
    int j = (int)(t % w);
    out[t] = in[m * Win + coff + j];
}

__global__ void flip_kernel(const float* __restrict__ in, float* __restrict__ out,
                            int B, int S, int C)
{
    long t = (long)blockIdx.x * blockDim.x + threadIdx.x;
    if (t >= (long)B * S * C) return;
    int c = (int)(t % C);
    long bs = t / C;
    int s = (int)(bs % S);
    int b = (int)(bs / S);
    out[((long)b * S + (S - 1 - s)) * C + c] = in[t];
}

// ---------------------------------------------------------------------------
// Farthest point sampling: one block per batch, sequential S iterations
// ---------------------------------------------------------------------------
__global__ void fps_kernel(const float* __restrict__ xyz, int* __restrict__ out_idx,
                           int n, int S)
{
    int b = blockIdx.x;
    const float* P = xyz + (long)b * n * 3;
    __shared__ float dists[2048];
    __shared__ float rmax[256];
    __shared__ int   rarg[256];
    __shared__ int   s_far;
    int tid = threadIdx.x;
    for (int i = tid; i < n; i += 256) dists[i] = 1e10f;
    if (tid == 0) s_far = 0;
    __syncthreads();
    for (int it = 0; it < S; ++it) {
        int far = s_far;
        if (tid == 0) out_idx[(long)b * S + it] = far;
        float cx = P[far * 3 + 0], cy = P[far * 3 + 1], cz = P[far * 3 + 2];
        float lm = -1.0f; int la = 0;
        for (int i = tid; i < n; i += 256) {
            float dx = P[i * 3 + 0] - cx, dy = P[i * 3 + 1] - cy, dz = P[i * 3 + 2] - cz;
            float nd = fminf(dists[i], dx * dx + dy * dy + dz * dz);
            dists[i] = nd;
            if (nd > lm) { lm = nd; la = i; }
        }
        rmax[tid] = lm; rarg[tid] = la;
        __syncthreads();
        for (int o = 128; o > 0; o >>= 1) {
            if (tid < o) {
                if (rmax[tid + o] > rmax[tid] ||
                    (rmax[tid + o] == rmax[tid] && rarg[tid + o] < rarg[tid])) {
                    rmax[tid] = rmax[tid + o]; rarg[tid] = rarg[tid + o];
                }
            }
            __syncthreads();
        }
        if (tid == 0) s_far = rarg[0];
        __syncthreads();
    }
}

__global__ void gather_kernel(const float* __restrict__ P, const int* __restrict__ idx,
                              float* __restrict__ out, int B, int n, int S, int C)
{
    long t = (long)blockIdx.x * blockDim.x + threadIdx.x;
    if (t >= (long)B * S * C) return;
    int c = (int)(t % C);
    long bs = t / C;
    int s = (int)(bs % S);
    int b = (int)(bs / S);
    out[t] = P[((long)b * n + idx[(long)b * S + s]) * C + c];
}

// K=24 nearest neighbors, one thread per query point (sorted ascending by d2)
__global__ void knn_kernel(const float* __restrict__ xyz, const float* __restrict__ qxyz,
                           int* __restrict__ knn, int B, int n, int S)
{
    long t = (long)blockIdx.x * blockDim.x + threadIdx.x;
    if (t >= (long)B * S) return;
    int b = (int)(t / S);
    const float* P = xyz + (long)b * n * 3;
    float qx = qxyz[t * 3 + 0], qy = qxyz[t * 3 + 1], qz = qxyz[t * 3 + 2];
    float bd[24]; int bi[24];
    #pragma unroll
    for (int j = 0; j < 24; ++j) { bd[j] = 3.4e38f; bi[j] = 0; }
    for (int i = 0; i < n; ++i) {
        float dx = P[i * 3 + 0] - qx, dy = P[i * 3 + 1] - qy, dz = P[i * 3 + 2] - qz;
        float d = dx * dx + dy * dy + dz * dz;
        if (d < bd[23]) {
            int j = 23;
            while (j > 0 && bd[j - 1] > d) { bd[j] = bd[j - 1]; bi[j] = bi[j - 1]; --j; }
            bd[j] = d; bi[j] = i;
        }
    }
    for (int j = 0; j < 24; ++j) knn[t * 24 + j] = bi[j];
}

// grouped deviation + per-batch sum / sumsq accumulation
__global__ void group_dev_kernel(const float* __restrict__ P, const float* __restrict__ NP,
                                 const int* __restrict__ knn, float* __restrict__ dev,
                                 float* __restrict__ sums, int B, int n, int S, int C, int K)
{
    long t = (long)blockIdx.x * blockDim.x + threadIdx.x;
    if (t >= (long)B * S * K * C) return;
    int c = (int)(t % C);
    long r = t / C;
    int k = (int)(r % K);
    long bs = r / K;
    int s = (int)(bs % S);
    int b = (int)(bs / S);
    float d = P[((long)b * n + knn[((long)b * S + s) * K + k]) * C + c]
            - NP[((long)b * S + s) * C + c];
    dev[t] = d;
    atomicAdd(&sums[b * 2 + 0], d);
    atomicAdd(&sums[b * 2 + 1], d * d);
}

// write Xg (b*S, 2C, K) channel-major: [norm-dev | anchor]
__global__ void group_write_kernel(const float* __restrict__ dev, const float* __restrict__ NP,
                                   const float* __restrict__ sums, const float* __restrict__ alpha,
                                   const float* __restrict__ beta, float* __restrict__ Xg,
                                   int B, int S, int C, int K)
{
    int C2 = 2 * C;
    long t = (long)blockIdx.x * blockDim.x + threadIdx.x;
    if (t >= (long)B * S * C2 * K) return;
    int k = (int)(t % K);
    long r = t / K;
    int c2 = (int)(r % C2);
    long bn = r / C2;
    int s = (int)(bn % S);
    int b = (int)(bn / S);
    float v;
    if (c2 < C) {
        float sm = sums[b * 2 + 0], sq = sums[b * 2 + 1];
        float cnt = (float)S * (float)K * (float)C;
        float var = (sq - sm * sm / cnt) / (cnt - 1.0f);
        float sd = sqrtf(fmaxf(var, 0.0f));
        v = dev[(((long)b * S + s) * K + k) * C + c2] / (sd + 1e-5f) * alpha[c2] + beta[c2];
    } else {
        v = NP[((long)b * S + s) * C + (c2 - C)];
    }
    Xg[t] = v;
}

// ---------------------------------------------------------------------------
// BatchNorm over the column dimension (per-channel stats)
// ---------------------------------------------------------------------------
__global__ void bn_stats_kernel(const float* __restrict__ Y, float* __restrict__ stats,
                                int O, long M, int S, int mode)
{
    int o = blockIdx.x;
    int tid = threadIdx.x;
    __shared__ float ssum[256], ssq[256];
    float s = 0.0f, q = 0.0f;
    for (long m = tid; m < M; m += 256) {
        long addr = (mode == 0)
            ? (long)(m / S) * O * S + (long)o * S + (m % S)
            : m * O + o;
        float v = Y[addr];
        s += v; q += v * v;
    }
    ssum[tid] = s; ssq[tid] = q;
    __syncthreads();
    for (int off = 128; off > 0; off >>= 1) {
        if (tid < off) { ssum[tid] += ssum[tid + off]; ssq[tid] += ssq[tid + off]; }
        __syncthreads();
    }
    if (tid == 0) {
        float mean = ssum[0] / (float)M;
        stats[o] = mean;
        stats[O + o] = ssq[0] / (float)M - mean * mean;
    }
}

__global__ void bn_apply_kernel(float* __restrict__ Y, const float* __restrict__ stats,
                                const float* __restrict__ g, const float* __restrict__ be,
                                const float* __restrict__ res,
                                int O, long M, int S, int mode, int relu)
{
    long t = (long)blockIdx.x * blockDim.x + threadIdx.x;
    if (t >= (long)O * M) return;
    int o = (int)(t / M);
    long m = t % M;
    long addr = (mode == 0)
        ? (long)(m / S) * O * S + (long)o * S + (m % S)
        : m * O + o;
    float v = Y[addr];
    v = (v - stats[o]) * rsqrtf(stats[O + o] + 1e-5f) * g[o] + be[o];
    if (res) v += res[addr];
    if (relu) v = fmaxf(v, 0.0f);
    Y[addr] = v;
}

// ---------------------------------------------------------------------------
// Attention pooling: softmax(x.w+b over S) weighted channel sum, block per bn
// mode0: (bn, C, S) channel-major; mode1: (bn, S, C) channels-last
// ---------------------------------------------------------------------------
__global__ void att_pool_kernel(const float* __restrict__ X, const float* __restrict__ w,
                                const float* __restrict__ bsc, float* __restrict__ out,
                                int C, int S, int mode)
{
    int bn = blockIdx.x;
    int tid = threadIdx.x;
    __shared__ float logits[512];
    __shared__ float red[256];
    const float* Xb = X + (long)bn * C * S;
    for (int s = tid; s < S; s += 256) {
        float acc = bsc[0];
        for (int c = 0; c < C; ++c) {
            float v = (mode == 0) ? Xb[(long)c * S + s] : Xb[(long)s * C + c];
            acc += v * w[c];
        }
        logits[s] = acc;
    }
    __syncthreads();
    float lm = -3.4e38f;
    for (int s = tid; s < S; s += 256) lm = fmaxf(lm, logits[s]);
    red[tid] = lm;
    __syncthreads();
    for (int o = 128; o > 0; o >>= 1) { if (tid < o) red[tid] = fmaxf(red[tid], red[tid + o]); __syncthreads(); }
    float mx = red[0];
    __syncthreads();
    float le = 0.0f;
    for (int s = tid; s < S; s += 256) { float e = __expf(logits[s] - mx); logits[s] = e; le += e; }
    red[tid] = le;
    __syncthreads();
    for (int o = 128; o > 0; o >>= 1) { if (tid < o) red[tid] += red[tid + o]; __syncthreads(); }
    float denom = red[0];
    __syncthreads();
    for (int c = tid; c < C; c += 256) {
        float acc = 0.0f;
        for (int s = 0; s < S; ++s) {
            float v = (mode == 0) ? Xb[(long)c * S + s] : Xb[(long)s * C + c];
            acc += logits[s] * v;
        }
        out[(long)bn * C + c] = acc / denom;
    }
}

// LayerNorm over channels-last C, one block per token
__global__ void ln_kernel(const float* __restrict__ X, const float* __restrict__ g,
                          const float* __restrict__ be, float* __restrict__ Y, int C)
{
    long t = blockIdx.x;
    const float* x = X + t * C;
    float* y = Y + t * C;
    int tid = threadIdx.x;
    __shared__ float red[256];
    float s = 0.0f;
    for (int c = tid; c < C; c += 256) s += x[c];
    red[tid] = s;
    __syncthreads();
    for (int o = 128; o > 0; o >>= 1) { if (tid < o) red[tid] += red[tid + o]; __syncthreads(); }
    float mean = red[0] / (float)C;
    __syncthreads();
    float q = 0.0f;
    for (int c = tid; c < C; c += 256) { float d = x[c] - mean; q += d * d; }
    red[tid] = q;
    __syncthreads();
    for (int o = 128; o > 0; o >>= 1) { if (tid < o) red[tid] += red[tid + o]; __syncthreads(); }
    float rstd = rsqrtf(red[0] / (float)C + 1e-5f);
    __syncthreads();
    for (int c = tid; c < C; c += 256) y[c] = (x[c] - mean) * rstd * g[c] + be[c];
}

// depthwise causal conv (width 4) + bias + silu, channels-last (B,S,Cin)
__global__ void cconv_silu_kernel(const float* __restrict__ X, const float* __restrict__ w,
                                  const float* __restrict__ bias, float* __restrict__ U,
                                  int B, int S, int Cin)
{
    long t = (long)blockIdx.x * blockDim.x + threadIdx.x;
    if (t >= (long)B * S * Cin) return;
    int ch = (int)(t % Cin);
    long bs = t / Cin;
    int s = (int)(bs % S);
    int b = (int)(bs / S);
    float acc = bias[ch];
    #pragma unroll
    for (int j = 0; j < 4; ++j) {
        int sp = s - 3 + j;
        if (sp >= 0) acc += X[((long)b * S + sp) * Cin + ch] * w[ch * 4 + j];
    }
    U[t] = acc / (1.0f + __expf(-acc));
}

// selective scan: thread per (batch, channel); 16-state kept in registers
__global__ void scan_kernel(const float* __restrict__ U, const float* __restrict__ Delta,
                            const float* __restrict__ dbc, const float* __restrict__ A_log,
                            const float* __restrict__ Dp, const float* __restrict__ Z,
                            float* __restrict__ Yo, int B, int S, int Din, int R)
{
    long t = (long)blockIdx.x * blockDim.x + threadIdx.x;
    if (t >= (long)B * Din) return;
    int ch = (int)(t % Din);
    int b = (int)(t / Din);
    float A[16], h[16];
    #pragma unroll
    for (int j = 0; j < 16; ++j) { A[j] = -__expf(A_log[ch * 16 + j]); h[j] = 0.0f; }
    float Dv = Dp[ch];
    int Wd = R + 32;
    for (int s = 0; s < S; ++s) {
        long base = (long)b * S + s;
        float dlt = Delta[base * Din + ch];
        float u = U[base * Din + ch];
        const float* bc = dbc + base * Wd + R;
        float y = 0.0f;
        #pragma unroll
        for (int j = 0; j < 16; ++j) {
            h[j] = __expf(dlt * A[j]) * h[j] + dlt * bc[j] * u;
            y += h[j] * bc[16 + j];
        }
        y += u * Dv;
        float z = Z[base * Din + ch];
        Yo[base * Din + ch] = y * (z / (1.0f + __expf(-z)));
    }
}

// ---------------------------------------------------------------------------
// Host orchestration
// ---------------------------------------------------------------------------
struct Lin2P { const float *b1, *b2, *be1, *be2, *g1, *g2, *w1, *w2; };
struct DirP  { const float *A_log, *D, *conv_b, *conv_w, *dt_b, *dt_w, *x_proj; };

extern "C" void kernel_launch(void* const* d_in, const int* in_sizes, int n_in,
                              void* d_out, int out_size, void* d_ws, size_t ws_size,
                              hipStream_t stream)
{
    (void)in_sizes; (void)n_in; (void)out_size; (void)ws_size;
    const int Bv = 16, Nv = 2048, Kn = 24;

    // Leaf order: d_in[0]=x; params flattened by sorted keys (jax tree order):
    // att1..4 (b,w): 1..8 | cls (b1,b2,be,g,w1,w2): 9..14 |
    // conv1:15 conv1_1:23 conv2:31 conv2_1:39 conv3:47 conv3_1:55 (b1,b2,be1,be2,g1,g2,w1,w2) |
    // embed (b,be,g,w): 63..66 | g0:67 g1:69 g2:71 (alpha,beta) |
    // m1:73 m2:91 m3:109 -> bwd(+0..6: A_log,D,conv_b,conv_w,dt_b,dt_w,x_proj),
    //                        fwd(+7..13), in_proj(+14), ln_b(+15), ln_g(+16), out_proj(+17)
    auto PF = [&](int i) -> const float* { return (const float*)d_in[i]; };
    auto L2 = [&](int b) -> Lin2P {
        return { PF(b), PF(b + 1), PF(b + 2), PF(b + 3), PF(b + 4), PF(b + 5), PF(b + 6), PF(b + 7) };
    };
    auto DIR = [&](int b) -> DirP {
        return { PF(b), PF(b + 1), PF(b + 2), PF(b + 3), PF(b + 4), PF(b + 5), PF(b + 6) };
    };

    // workspace arena
    char* base = (char*)d_ws;
    size_t off = 0;
    auto alloc = [&](size_t bytes) -> void* {
        void* p = base + off;
        off = (off + bytes + 255) & ~(size_t)255;
        return p;
    };
    auto allocf = [&](long n) -> float* { return (float*)alloc((size_t)n * sizeof(float)); };
    auto alloci = [&](long n) -> int*   { return (int*)alloc((size_t)n * sizeof(int)); };
    auto nb = [](long n) -> int { return (int)((n + 255) / 256); };

    auto gemm = [&](const float* W, const float* X, const float* bias, float* Y,
                    int O, int Cin, int M, int S, int inm, int outm, int act) {
        dim3 g((unsigned)((M + 63) / 64), (unsigned)((O + 63) / 64));
        wmma_gemm_kernel<<<g, 256, 0, stream>>>(W, X, bias, Y, O, Cin, M, S, inm, outm, act);
    };

    float* stats = allocf(2 * 1024);
    auto bn_relu = [&](float* Y, const float* g, const float* be, const float* res,
                       int O, int M, int S, int mode, int relu) {
        bn_stats_kernel<<<O, 256, 0, stream>>>(Y, stats, O, (long)M, S, mode);
        bn_apply_kernel<<<nb((long)O * M), 256, 0, stream>>>(Y, stats, g, be, res,
                                                             O, (long)M, S, mode, relu);
    };
    auto lin2 = [&](const float* X, float* H, float* Y, Lin2P p, int C, int M, int S, int mode) {
        gemm(p.w1, X, p.b1, H, C / 2, C, M, S, mode, mode, 0);
        bn_relu(H, p.g1, p.be1, nullptr, C / 2, M, S, mode, 1);
        gemm(p.w2, H, p.b2, Y, C, C / 2, M, S, mode, mode, 0);
        bn_relu(Y, p.g2, p.be2, X, C, M, S, mode, 1);
    };

    auto mamba_dir = [&](const float* xi, const float* z, DirP p,
                         int Sv, int d, int din, int R, float* yout) {
        (void)d;
        long toks = (long)Bv * Sv;
        float* u = allocf(toks * din);
        cconv_silu_kernel<<<nb(toks * din), 256, 0, stream>>>(xi, p.conv_w, p.conv_b, u, Bv, Sv, din);
        float* dbc = allocf(toks * (R + 32));
        gemm(p.x_proj, u, nullptr, dbc, R + 32, din, (int)toks, 1, 1, 1, 0);
        float* dt = allocf(toks * R);
        split_kernel<<<nb(toks * R), 256, 0, stream>>>(dbc, dt, toks, R + 32, 0, R);
        float* delta = allocf(toks * din);
        gemm(p.dt_w, dt, p.dt_b, delta, din, R, (int)toks, 1, 1, 1, 3); // softplus
        scan_kernel<<<nb((long)Bv * din), 256, 0, stream>>>(u, delta, dbc, p.A_log, p.D, z,
                                                            yout, Bv, Sv, din, R);
    };

    auto mamba = [&](float* f, int Sv, int d, int mb) {
        int din = 2 * d, R = (d + 15) / 16;
        long toks = (long)Bv * Sv;
        float* h = allocf(toks * d);
        ln_kernel<<<(unsigned)toks, 256, 0, stream>>>(f, PF(mb + 16), PF(mb + 15), h, d);
        float* xz = allocf(toks * 2 * din);
        gemm(PF(mb + 14), h, nullptr, xz, 2 * din, d, (int)toks, 1, 1, 1, 0);
        float* xi = allocf(toks * din);
        float* z  = allocf(toks * din);
        split_kernel<<<nb(toks * din), 256, 0, stream>>>(xz, xi, toks, 2 * din, 0, din);
        split_kernel<<<nb(toks * din), 256, 0, stream>>>(xz, z,  toks, 2 * din, din, din);
        float* yf = allocf(toks * din);
        mamba_dir(xi, z, DIR(mb + 7), Sv, d, din, R, yf);
        float* xir = allocf(toks * din);
        float* zr  = allocf(toks * din);
        flip_kernel<<<nb(toks * din), 256, 0, stream>>>(xi, xir, Bv, Sv, din);
        flip_kernel<<<nb(toks * din), 256, 0, stream>>>(z,  zr,  Bv, Sv, din);
        float* ybr = allocf(toks * din);
        mamba_dir(xir, zr, DIR(mb), Sv, d, din, R, ybr);
        float* yb = allocf(toks * din);
        flip_kernel<<<nb(toks * din), 256, 0, stream>>>(ybr, yb, Bv, Sv, din);
        add_kernel<<<nb(toks * din), 256, 0, stream>>>(yf, yb, yf, toks * din);
        float* o = allocf(toks * d);
        gemm(PF(mb + 17), yf, nullptr, o, d, din, (int)toks, 1, 1, 1, 0);
        add_kernel<<<nb(toks * d), 256, 0, stream>>>(f, o, f, toks * d);
    };

    auto grouper = [&](const float* xyz_in, const float* pts_in, int n_cur, int Snew, int C,
                       const float* alpha, const float* beta, float* new_xyz, float** Xg_out) {
        int* fi = alloci((long)Bv * Snew);
        fps_kernel<<<Bv, 256, 0, stream>>>(xyz_in, fi, n_cur, Snew);
        gather_kernel<<<nb((long)Bv * Snew * 3), 256, 0, stream>>>(xyz_in, fi, new_xyz, Bv, n_cur, Snew, 3);
        float* np = allocf((long)Bv * Snew * C);
        gather_kernel<<<nb((long)Bv * Snew * C), 256, 0, stream>>>(pts_in, fi, np, Bv, n_cur, Snew, C);
        int* kn = alloci((long)Bv * Snew * Kn);
        knn_kernel<<<nb((long)Bv * Snew), 256, 0, stream>>>(xyz_in, new_xyz, kn, Bv, n_cur, Snew);
        float* dev = allocf((long)Bv * Snew * Kn * C);
        float* sums = allocf(2 * Bv);
        zero_kernel<<<1, 64, 0, stream>>>(sums, 2 * Bv);
        group_dev_kernel<<<nb((long)Bv * Snew * Kn * C), 256, 0, stream>>>(
            pts_in, np, kn, dev, sums, Bv, n_cur, Snew, C, Kn);
        float* Xg = allocf((long)Bv * Snew * 2 * C * Kn);
        group_write_kernel<<<nb((long)Bv * Snew * 2 * C * Kn), 256, 0, stream>>>(
            dev, np, sums, alpha, beta, Xg, Bv, Snew, C, Kn);
        *Xg_out = Xg;
    };

    // persistent buffers
    float* xyz0 = allocf((long)Bv * Nv * 3);
    float* xyz1 = allocf((long)Bv * 512 * 3);
    float* xyz2 = allocf((long)Bv * 256 * 3);
    float* xyz3 = allocf((long)Bv * 128 * 3);
    float* f1  = allocf((long)Bv * 512 * 64);
    float* f1b = allocf((long)Bv * 512 * 64);
    float* f2  = allocf((long)Bv * 256 * 128);
    float* f2b = allocf((long)Bv * 256 * 128);
    float* f3  = allocf((long)Bv * 128 * 256);
    float* f3b = allocf((long)Bv * 128 * 256);
    float* pooled = allocf((long)Bv * 256);

    transpose_in_kernel<<<nb((long)Bv * Nv * 3), 256, 0, stream>>>(PF(0), xyz0, Bv, Nv);

    size_t mark = off;

    // ---------------- Stage 1: 2048 -> 512, C 3 -> 64 ----------------
    {
        off = mark;
        float* Xg;
        grouper(xyz0, xyz0, Nv, 512, 3, PF(67), PF(68), xyz1, &Xg);
        int M1 = Bv * 512 * Kn;
        float* Y0 = allocf((long)64 * M1);
        gemm(PF(66), Xg, PF(63), Y0, 64, 6, M1, Kn, 0, 0, 0);       // embed 6->64
        bn_relu(Y0, PF(65), PF(64), nullptr, 64, M1, Kn, 0, 1);
        float* Hb = allocf((long)32 * M1);
        float* Y1 = allocf((long)64 * M1);
        lin2(Y0, Hb, Y1, L2(15), 64, M1, Kn, 0);                    // conv1
        att_pool_kernel<<<Bv * 512, 256, 0, stream>>>(Y1, PF(2), PF(1), f1, 64, Kn, 0);
        off = mark;
        mamba(f1, 512, 64, 73);                                     // m1
        off = mark;
        float* Hc = allocf((long)32 * (Bv * 512));
        lin2(f1, Hc, f1b, L2(23), 64, Bv * 512, 1, 1);              // conv1_1
    }

    // ---------------- Stage 2: 512 -> 256, C 64 -> 128 ----------------
    {
        off = mark;
        float* Xg;
        grouper(xyz1, f1b, 512, 256, 64, PF(69), PF(70), xyz2, &Xg);
        int M2 = Bv * 256 * Kn;
        float* H2 = allocf((long)64 * M2);
        float* Y2 = allocf((long)128 * M2);
        lin2(Xg, H2, Y2, L2(31), 128, M2, Kn, 0);                   // conv2
        att_pool_kernel<<<Bv * 256, 256, 0, stream>>>(Y2, PF(4), PF(3), f2, 128, Kn, 0);
        off = mark;
        mamba(f2, 256, 128, 91);                                    // m2
        off = mark;
        float* Hc = allocf((long)64 * (Bv * 256));
        lin2(f2, Hc, f2b, L2(39), 128, Bv * 256, 1, 1);             // conv2_1
    }

    // ---------------- Stage 3: 256 -> 128, C 128 -> 256 ----------------
    {
        off = mark;
        float* Xg;
        grouper(xyz2, f2b, 256, 128, 128, PF(71), PF(72), xyz3, &Xg);
        int M3 = Bv * 128 * Kn;
        float* H3 = allocf((long)128 * M3);
        float* Y3 = allocf((long)256 * M3);
        lin2(Xg, H3, Y3, L2(47), 256, M3, Kn, 0);                   // conv3
        att_pool_kernel<<<Bv * 128, 256, 0, stream>>>(Y3, PF(6), PF(5), f3, 256, Kn, 0);
        off = mark;
        mamba(f3, 128, 256, 109);                                   // m3
        off = mark;
        float* Hc = allocf((long)128 * (Bv * 128));
        lin2(f3, Hc, f3b, L2(55), 256, Bv * 128, 1, 1);             // conv3_1
    }

    // ---------------- Head: att4 pool + classifier ----------------
    {
        off = mark;
        att_pool_kernel<<<Bv, 256, 0, stream>>>(f3b, PF(8), PF(7), pooled, 256, 128, 1);
        float* Hc1 = allocf((long)Bv * 256);
        gemm(PF(13), pooled, PF(9), Hc1, 256, 256, Bv, 1, 1, 1, 0);
        bn_relu(Hc1, PF(12), PF(11), nullptr, 256, Bv, 1, 1, 1);
        gemm(PF(14), Hc1, PF(10), (float*)d_out, 6, 256, Bv, 1, 1, 1, 0);
    }
}
